// Transformer_81209241633224
// MI455X (gfx1250) — compile-verified, matrix operand order checked by reading, not run on hardware
//
#include <hip/hip_runtime.h>

typedef __bf16 bf16_t;
typedef bf16_t bf16x8  __attribute__((ext_vector_type(8)));
typedef bf16_t bf16x16 __attribute__((ext_vector_type(16)));
typedef float  f32x8   __attribute__((ext_vector_type(8)));
typedef unsigned int u32x4 __attribute__((ext_vector_type(4)));
typedef int i32x4 __attribute__((ext_vector_type(4)));
typedef int i32x8 __attribute__((ext_vector_type(8)));

#define N_CTX   8192
#define N_IN    2048
#define N_STATE 512
#define D_HEAD  64
#define QKV_N   1536
#define N_DEPTH 6

static __device__ __forceinline__ bf16x16 load_frag(const bf16_t* p) {
  bf16x8 lo = *(const bf16x8*)p;
  bf16x8 hi = *(const bf16x8*)(p + 8);
  bf16x16 f;
#pragma unroll
  for (int e = 0; e < 8; ++e) { f[e] = lo[e]; f[e + 8] = hi[e]; }
  return f;
}

static __device__ __forceinline__ f32x8 wmma_bf16(bf16x16 a, bf16x16 b, f32x8 c) {
  return __builtin_amdgcn_wmma_f32_16x16x32_bf16(false, a, false, b, (short)0, c,
                                                 false, false);
}

// Tensor Data Mover: DMA one 128-row x 64-byte tile (row pitch K*2 bytes) from
// global into LDS with HW padding 64B->+16B, matching the [128][40] bf16 layout.
// D# per CDNA5 ISA: data_size=8B, tile_dim0=8, tile_dim1=128,
// tensor_dim0_stride=K*2/8, pad_interval code 3 (16 dwords), pad_amount code 3
// (4 dwords), group0 type=2, count=1.
static __device__ __forceinline__ void tdm_load_tile(unsigned lds_addr,
                                                     const bf16_t* gptr,
                                                     int kElems) {
  unsigned long long ga = (unsigned long long)(uintptr_t)gptr;
  u32x4 g0;
  g0[0] = 1u;                                           // count=1 (valid)
  g0[1] = lds_addr;                                     // lds_addr (bytes)
  g0[2] = (unsigned)(ga & 0xFFFFFFFFull);               // global_addr[31:0]
  g0[3] = (unsigned)((ga >> 32) & 0x1FFFFFFull) | (2u << 30);  // addr[56:32]|type=2
  i32x8 g1;
  g1[0] = (3 << 16) | (1 << 20) | (3 << 22) | (3 << 25); // data_size=8B, pad en,
                                                         // interval=16dw, amt=4dw
  g1[1] = (8 << 16);        // tensor_dim0[15:0] = 8 (8B units) at bits 63:48
  g1[2] = (128 << 16);      // tensor_dim0 hi=0 | tensor_dim1[15:0]=128
  g1[3] = (8 << 16);        // tensor_dim1 hi=0 | tile_dim0=8
  g1[4] = 128;              // tile_dim1=128 | tile_dim2=0
  g1[5] = kElems >> 2;      // tensor_dim0_stride = K*2/8 (8B units), low 32
  g1[6] = 0;                // stride0 hi | stride1 lo
  g1[7] = 0;                // stride1 hi
  i32x4 gz = {0, 0, 0, 0};
#if __clang_major__ >= 23
  i32x8 gz8 = {0, 0, 0, 0, 0, 0, 0, 0};
  __builtin_amdgcn_tensor_load_to_lds(g0, g1, gz, gz, gz8, 0);
#else
  __builtin_amdgcn_tensor_load_to_lds(g0, g1, gz, gz, 0);
#endif
}

// ---------------------------------------------------------------- utilities

__global__ __launch_bounds__(256) void copy_kernel(const float4* __restrict__ in,
                                                   float4* __restrict__ out, int n4) {
  int i = blockIdx.x * 256 + threadIdx.x;
  if (i < n4) out[i] = in[i];
}

// W (K x N, f32, row-major) -> Wt (N x K, bf16, row-major)
__global__ __launch_bounds__(256) void convT_kernel(const float* __restrict__ W,
                                                    bf16_t* __restrict__ Wt,
                                                    int K, int N) {
  __shared__ float t[32][33];
  const int k0 = blockIdx.x * 32, n0 = blockIdx.y * 32;
  const int tx = threadIdx.x & 31, ty = threadIdx.x >> 5;
#pragma unroll
  for (int i = ty; i < 32; i += 8) t[i][tx] = W[(size_t)(k0 + i) * N + n0 + tx];
  __syncthreads();
#pragma unroll
  for (int i = ty; i < 32; i += 8)
    Wt[(size_t)(n0 + i) * K + k0 + tx] = (bf16_t)t[tx][i];
}

// LayerNorm over last dim (2048) of f32 input -> bf16 output
__global__ __launch_bounds__(256) void ln_kernel(const float* __restrict__ x,
                                                 const float* __restrict__ g,
                                                 const float* __restrict__ b,
                                                 bf16_t* __restrict__ out) {
  __shared__ float red[256];
  const int row = blockIdx.x;
  const int tid = threadIdx.x;
  const float* xr = x + (size_t)row * N_IN;
  float local[8];
  float s = 0.f;
#pragma unroll
  for (int i = 0; i < 8; ++i) { local[i] = xr[tid + i * 256]; s += local[i]; }
  red[tid] = s; __syncthreads();
  for (int off = 128; off; off >>= 1) {
    if (tid < off) red[tid] += red[tid + off];
    __syncthreads();
  }
  const float mean = red[0] * (1.f / N_IN);
  __syncthreads();
  float s2 = 0.f;
#pragma unroll
  for (int i = 0; i < 8; ++i) { float d = local[i] - mean; s2 += d * d; }
  red[tid] = s2; __syncthreads();
  for (int off = 128; off; off >>= 1) {
    if (tid < off) red[tid] += red[tid + off];
    __syncthreads();
  }
  const float rstd = rsqrtf(red[0] * (1.f / N_IN) + 1e-5f);
#pragma unroll
  for (int i = 0; i < 8; ++i) {
    int c = tid + i * 256;
    out[(size_t)row * N_IN + c] = (bf16_t)((local[i] - mean) * rstd * g[c] + b[c]);
  }
}

// ------------------------------------------------------------------- GEMM
// C[M,N] = A[M,K](bf16) * Bt[N,K]^T(bf16) + bias, fused epilogues:
//   epi 0: -> bf16          epi 1: +resid -> f32        epi 2: quickGELU -> bf16
// Tile staging is done by the Tensor Data Mover (one issuing wave), double
// buffered; compute waves only do ds_load fragments + v_wmma.
__global__ __launch_bounds__(256, 2) void gemm_bf16_kernel(
    const bf16_t* __restrict__ A, const bf16_t* __restrict__ Bt,
    const float* __restrict__ bias, const float* __restrict__ resid,
    float* __restrict__ outF, bf16_t* __restrict__ outH,
    int M, int N, int K, int epi) {
  __shared__ __align__(16) bf16_t sA[2][128][40];
  __shared__ __align__(16) bf16_t sB[2][128][40];
  const int tid = threadIdx.x;
  const int lane = tid & 31;
  const int w = tid >> 5;
  const int wm = w >> 1, wn = w & 1;
  const int tileM = blockIdx.y * 128;
  const int tileN = blockIdx.x * 128;

  const f32x8 zero8 = {0.f, 0.f, 0.f, 0.f, 0.f, 0.f, 0.f, 0.f};
  f32x8 acc[2][4];
#pragma unroll
  for (int i = 0; i < 2; ++i)
#pragma unroll
    for (int j = 0; j < 4; ++j) acc[i][j] = zero8;

  const bf16_t* gA = A + (size_t)tileM * K;
  const bf16_t* gB = Bt + (size_t)tileN * K;
  const int nk = K >> 5;

  if (w == 0) {
    tdm_load_tile((unsigned)(uintptr_t)&sA[0][0][0], gA, K);
    tdm_load_tile((unsigned)(uintptr_t)&sB[0][0][0], gB, K);
  }

  for (int kt = 0; kt < nk; ++kt) {
    const int cur = kt & 1;
    __builtin_amdgcn_s_wait_tensorcnt(0);  // wave0: tiles for `cur` landed
    __syncthreads();                       // publish LDS to all waves
    if (w == 0 && (kt + 1) < nk) {         // prefetch next tiles via TDM
      const int nxt = cur ^ 1;
      tdm_load_tile((unsigned)(uintptr_t)&sA[nxt][0][0], gA + ((kt + 1) << 5), K);
      tdm_load_tile((unsigned)(uintptr_t)&sB[nxt][0][0], gB + ((kt + 1) << 5), K);
    }
    bf16x16 af[2], bfr[4];
#pragma unroll
    for (int i = 0; i < 2; ++i)
      af[i] = load_frag(&sA[cur][wm * 32 + i * 16 + (lane & 15)][(lane >> 4) * 16]);
#pragma unroll
    for (int j = 0; j < 4; ++j)
      bfr[j] = load_frag(&sB[cur][wn * 64 + j * 16 + (lane & 15)][(lane >> 4) * 16]);
#pragma unroll
    for (int i = 0; i < 2; ++i)
#pragma unroll
      for (int j = 0; j < 4; ++j) acc[i][j] = wmma_bf16(af[i], bfr[j], acc[i][j]);
  }

#pragma unroll
  for (int i = 0; i < 2; ++i)
#pragma unroll
    for (int j = 0; j < 4; ++j) {
      const int nb = tileN + wn * 64 + j * 16 + (lane & 15);
      const float bv = bias[nb];
#pragma unroll
      for (int r = 0; r < 8; ++r) {
        const int m = tileM + wm * 32 + i * 16 + r + ((lane >> 4) << 3);
        const size_t idx = (size_t)m * N + nb;
        float v = acc[i][j][r] + bv;
        if (epi == 1) {
          outF[idx] = v + resid[idx];
        } else if (epi == 2) {
          outH[idx] = (bf16_t)(v / (1.f + __expf(-1.702f * v)));
        } else {
          outH[idx] = (bf16_t)v;
        }
      }
    }
}

// --------------------------------------------------------------- attention
// FUNC 1: block attn (causal within 128-block)
// FUNC 2: transpose block attn (causal across 64 blocks, stride 128)
// FUNC 3: prev-block attn (no mask; block 0 -> zeros)
template <int FUNC>
__global__ __launch_bounds__(256) void attn_kernel(const bf16_t* __restrict__ qkv,
                                                   bf16_t* __restrict__ aout) {
  constexpr int M = (FUNC == 2) ? 64 : 128;   // queries per group
  constexpr int NK = M;                       // keys per group
  constexpr int STEP = (FUNC == 2) ? 128 : 1; // seq stride between rows
  constexpr bool CAUSAL = (FUNC != 3);
  constexpr int NSUB = NK / 16;
  constexpr int NWAV = M / 16;

  __shared__ __align__(16) bf16_t P[M][NK + 8];
  __shared__ __align__(16) bf16_t Vt[D_HEAD][NK + 8];

  const int grp = blockIdx.x, h = blockIdx.y;
  const int tid = threadIdx.x, lane = tid & 31, w = tid >> 5;
  const int hoff = h * D_HEAD;

  int qbase, kbase;
  if (FUNC == 2) { qbase = grp; kbase = grp; }
  else           { qbase = grp * 128; kbase = (FUNC == 3) ? (grp - 1) * 128 : grp * 128; }

  if (FUNC == 3 && grp == 0) {  // k=v=0 -> output exactly zero
    for (int idx = tid; idx < 128 * D_HEAD; idx += 256)
      aout[(size_t)(qbase + (idx >> 6)) * N_STATE + hoff + (idx & 63)] = (bf16_t)0.f;
    return;
  }

  // Stage V transposed: Vt[d][key]  (key contiguous -> B-fragment layout)
  for (int idx = tid; idx < NK * D_HEAD; idx += 256) {
    const int key = idx >> 6, d = idx & 63;
    Vt[d][key] = qkv[(size_t)(kbase + key * STEP) * QKV_N + 2 * N_STATE + hoff + d];
  }

  const f32x8 zero8 = {0.f, 0.f, 0.f, 0.f, 0.f, 0.f, 0.f, 0.f};

  if (w < NWAV) {
    f32x8 s[NSUB];
#pragma unroll
    for (int j = 0; j < NSUB; ++j) s[j] = zero8;

    // S = Q K^T  (d = 64 -> two k-steps of 32)
#pragma unroll
    for (int ks = 0; ks < D_HEAD / 32; ++ks) {
      const int qi = w * 16 + (lane & 15);
      bf16x16 af = load_frag(qkv + (size_t)(qbase + qi * STEP) * QKV_N + hoff +
                             ks * 32 + ((lane >> 4) * 16));
#pragma unroll
      for (int j = 0; j < NSUB; ++j) {
        const int kj = j * 16 + (lane & 15);
        bf16x16 bfr = load_frag(qkv + (size_t)(kbase + kj * STEP) * QKV_N + N_STATE +
                                hoff + ks * 32 + ((lane >> 4) * 16));
        s[j] = wmma_bf16(af, bfr, s[j]);
      }
    }

    // row softmax (scale 1/sqrt(64)); row lives in register r across 16 lanes
#pragma unroll
    for (int r = 0; r < 8; ++r) {
      const int ml = r + ((lane >> 4) << 3);
      const int qi = w * 16 + ml;
      float sv[NSUB];
      float mx = -1e30f;
#pragma unroll
      for (int j = 0; j < NSUB; ++j) {
        const int kj = j * 16 + (lane & 15);
        float v = s[j][r] * 0.125f;
        if (CAUSAL && kj > qi) v = -1e9f;
        sv[j] = v;
        mx = fmaxf(mx, v);
      }
#pragma unroll
      for (int msk = 1; msk <= 8; msk <<= 1) mx = fmaxf(mx, __shfl_xor(mx, msk, 32));
      float sum = 0.f;
#pragma unroll
      for (int j = 0; j < NSUB; ++j) { sv[j] = __expf(sv[j] - mx); sum += sv[j]; }
#pragma unroll
      for (int msk = 1; msk <= 8; msk <<= 1) sum += __shfl_xor(sum, msk, 32);
      const float inv = 1.f / sum;
#pragma unroll
      for (int j = 0; j < NSUB; ++j)
        P[qi][j * 16 + (lane & 15)] = (bf16_t)(sv[j] * inv);
    }
  }
  __syncthreads();

  if (w < NWAV) {
    f32x8 o[4];
#pragma unroll
    for (int j = 0; j < 4; ++j) o[j] = zero8;
#pragma unroll
    for (int kt = 0; kt < NK / 32; ++kt) {
      const int qi = w * 16 + (lane & 15);
      bf16x16 af = load_frag(&P[qi][kt * 32 + ((lane >> 4) * 16)]);
#pragma unroll
      for (int j = 0; j < 4; ++j) {
        bf16x16 bfr = load_frag(&Vt[j * 16 + (lane & 15)][kt * 32 + ((lane >> 4) * 16)]);
        o[j] = wmma_bf16(af, bfr, o[j]);
      }
    }
#pragma unroll
    for (int j = 0; j < 4; ++j)
#pragma unroll
      for (int r = 0; r < 8; ++r) {
        const int ml = r + ((lane >> 4) << 3);
        aout[(size_t)(qbase + (w * 16 + ml) * STEP) * N_STATE + hoff + j * 16 +
             (lane & 15)] = (bf16_t)o[j][r];
      }
  }
}

// ------------------------------------------------------------------- host

extern "C" void kernel_launch(void* const* d_in, const int* in_sizes, int n_in,
                              void* d_out, int out_size, void* d_ws, size_t ws_size,
                              hipStream_t stream) {
  (void)in_sizes; (void)n_in; (void)out_size; (void)ws_size;
  const float* x_in  = (const float*)d_in[0];
  const float* ln0_g = (const float*)d_in[1];
  const float* ln0_b = (const float*)d_in[2];
  const float* qkv_w = (const float*)d_in[3];
  const float* qkv_b = (const float*)d_in[4];
  const float* ap_w  = (const float*)d_in[5];
  const float* ap_b  = (const float*)d_in[6];
  const float* ln1_g = (const float*)d_in[7];
  const float* ln1_b = (const float*)d_in[8];
  const float* fc_w  = (const float*)d_in[9];
  const float* fc_b  = (const float*)d_in[10];
  const float* mp_w  = (const float*)d_in[11];
  const float* mp_b  = (const float*)d_in[12];
  float* xout = (float*)d_out;

  char* ws = (char*)d_ws;
  size_t off = 0;
  auto alloc = [&](size_t bytes) -> void* {
    void* p = ws + off;
    off = (off + bytes + 255) & ~(size_t)255;
    return p;
  };
  bf16_t* hbuf   = (bf16_t*)alloc((size_t)N_CTX * N_IN * 2);
  bf16_t* qkvbuf = (bf16_t*)alloc((size_t)N_CTX * QKV_N * 2);
  bf16_t* abuf   = (bf16_t*)alloc((size_t)N_CTX * N_STATE * 2);
  float*  xabuf  = (float*)alloc((size_t)N_CTX * N_IN * 4);
  bf16_t* mbuf   = (bf16_t*)alloc((size_t)N_CTX * N_IN * 2);
  bf16_t* wqkvT  = (bf16_t*)alloc((size_t)QKV_N * N_IN * 2);
  bf16_t* wapT   = (bf16_t*)alloc((size_t)N_IN * N_STATE * 2);
  bf16_t* wfcT   = (bf16_t*)alloc((size_t)N_IN * N_IN * 2);
  bf16_t* wmpT   = (bf16_t*)alloc((size_t)N_IN * N_IN * 2);

  // running activation x lives in d_out; re-init every launch (deterministic)
  const int n4 = (N_CTX * N_IN) / 4;
  copy_kernel<<<(n4 + 255) / 256, 256, 0, stream>>>((const float4*)x_in,
                                                    (float4*)xout, n4);

  for (int d = 0; d < N_DEPTH; ++d) {
    const int func = (d % 3) + 1;  // [1,2,3][d%3]

    convT_kernel<<<dim3(N_IN / 32, QKV_N / 32), 256, 0, stream>>>(
        qkv_w + (size_t)d * N_IN * QKV_N, wqkvT, N_IN, QKV_N);
    convT_kernel<<<dim3(N_STATE / 32, N_IN / 32), 256, 0, stream>>>(
        ap_w + (size_t)d * N_STATE * N_IN, wapT, N_STATE, N_IN);
    convT_kernel<<<dim3(N_IN / 32, N_IN / 32), 256, 0, stream>>>(
        fc_w + (size_t)d * N_IN * N_IN, wfcT, N_IN, N_IN);
    convT_kernel<<<dim3(N_IN / 32, N_IN / 32), 256, 0, stream>>>(
        mp_w + (size_t)d * N_IN * N_IN, wmpT, N_IN, N_IN);

    ln_kernel<<<N_CTX, 256, 0, stream>>>(xout, ln0_g + (size_t)d * N_IN,
                                         ln0_b + (size_t)d * N_IN, hbuf);

    gemm_bf16_kernel<<<dim3(QKV_N / 128, N_CTX / 128), 256, 0, stream>>>(
        hbuf, wqkvT, qkv_b + (size_t)d * QKV_N, nullptr, nullptr, qkvbuf,
        N_CTX, QKV_N, N_IN, 0);

    if (func == 1)
      attn_kernel<1><<<dim3(64, 8), 256, 0, stream>>>(qkvbuf, abuf);
    else if (func == 2)
      attn_kernel<2><<<dim3(128, 8), 256, 0, stream>>>(qkvbuf, abuf);
    else
      attn_kernel<3><<<dim3(64, 8), 256, 0, stream>>>(qkvbuf, abuf);

    gemm_bf16_kernel<<<dim3(N_IN / 128, N_CTX / 128), 256, 0, stream>>>(
        abuf, wapT, ap_b + (size_t)d * N_IN, xout, xabuf, nullptr,
        N_CTX, N_IN, N_STATE, 1);

    ln_kernel<<<N_CTX, 256, 0, stream>>>(xabuf, ln1_g + (size_t)d * N_IN,
                                         ln1_b + (size_t)d * N_IN, hbuf);

    gemm_bf16_kernel<<<dim3(N_IN / 128, N_CTX / 128), 256, 0, stream>>>(
        hbuf, wfcT, fc_b + (size_t)d * N_IN, nullptr, nullptr, mbuf,
        N_CTX, N_IN, N_IN, 2);

    gemm_bf16_kernel<<<dim3(N_IN / 128, N_CTX / 128), 256, 0, stream>>>(
        mbuf, wmpT, mp_b + (size_t)d * N_IN, xabuf, xout, nullptr,
        N_CTX, N_IN, N_IN, 1);
  }
}
